// Transformer_16140487098459
// MI455X (gfx1250) — matrix operand order repaired and checked by reading, not compiled, not run on hardware
//
#include <hip/hip_runtime.h>
#include <hip/hip_bf16.h>

// ---------------------------------------------------------------------------
// Transformer block for MI455X (gfx1250, wave32, WMMA f16 -> f32 accumulate)
// B=4 S=2048 D=512 H=8 dk=64
// ---------------------------------------------------------------------------

typedef __attribute__((ext_vector_type(16))) _Float16 v16h;
typedef __attribute__((ext_vector_type(8)))  _Float16 v8h;
typedef __attribute__((ext_vector_type(8)))  float    v8f;

#define B_    4
#define S_    2048
#define D_    512
#define H_    8
#define DK_   64
#define DFF_  2048
#define ROWS_ (B_ * S_)   // 8192

#define LOG2E_ 1.4426950408889634f

// D = A(16x32) * B(32x16) + C, f16 inputs, f32 accumulate
__device__ __forceinline__ v8f wmma16(v16h a, v16h b, v8f c) {
  return __builtin_amdgcn_wmma_f32_16x16x32_f16(false, a, false, b, (short)0, c,
                                                false, false);
}

// A-fragment (16x32 f16). This code's own k order (not the ISA table order): lane (m=lane&15, hi=lane>>4) holds
// K = koff + hi*16 + {0..15} (contiguous, matching B-fragments).
__device__ __forceinline__ v16h afrag_h(const _Float16* row, int koff, int hi) {
  v8h lo = *(const v8h*)(row + koff + hi * 16);
  v8h hh = *(const v8h*)(row + koff + hi * 16 + 8);
  v16h a;
#pragma unroll
  for (int t = 0; t < 8; ++t) { a[t] = lo[t]; a[8 + t] = hh[t]; }
  return a;
}

// Same A-fragment but sourced from f32 with scale, converting to f16.
__device__ __forceinline__ v16h afrag_f32(const float* row, int koff, int hi,
                                          float scale) {
  const float* p0 = row + koff + hi * 16;
  const float* p1 = row + koff + hi * 16 + 8;
  v16h a;
#pragma unroll
  for (int t = 0; t < 8; ++t) {
    a[t]     = (_Float16)(p0[t] * scale);
    a[8 + t] = (_Float16)(p1[t] * scale);
  }
  return a;
}

// B-fragment (32x16 f16): lane (n=lane&15, hi=lane>>4) holds col n,
// K = hi*16 + {0..15} -> 16 contiguous halves when source is K-contiguous.
__device__ __forceinline__ v16h bfrag_h(const _Float16* p) {
  return *(const v16h*)p;
}

// ---------------------------------------------------------------------------
// 1) QKV projection: per-head dk x dk GEMM, f32 -> f16.
//    gridDim.x = ROWS_/16 (16-row tiles), gridDim.y = {0:q,1:k,2:v}
//    256 threads = 8 waves, wave w handles head w.
//    Q scaled by log2e/sqrt(dk) so attention softmax runs in exp2 domain.
//    V written transposed [B,H,dk,S].
// ---------------------------------------------------------------------------
__global__ void qkv_proj_kernel(const float* __restrict__ query,
                                const float* __restrict__ key,
                                const float* __restrict__ value,
                                const float* __restrict__ wq,
                                const float* __restrict__ wk,
                                const float* __restrict__ wv,
                                _Float16* __restrict__ Qh,
                                _Float16* __restrict__ Kh,
                                _Float16* __restrict__ Vt) {
  __shared__ __align__(32) _Float16 wT[DK_ * DK_];  // transposed weight, f16
  const int which = blockIdx.y;
  const float* w   = (which == 0) ? wq : (which == 1) ? wk : wv;
  const float* inp = (which == 0) ? query : (which == 1) ? key : value;

  // Stage weight transposed: wT[e*64 + d] = w[d*64 + e]
  for (int i = threadIdx.x; i < DK_ * DK_; i += blockDim.x) {
    int d = i >> 6, e = i & 63;
    wT[e * DK_ + d] = (_Float16)w[i];
  }
  __syncthreads();

  const int wave = threadIdx.x >> 5;
  const int lane = threadIdx.x & 31;
  const int h    = wave;                 // head
  const int row0 = blockIdx.x * 16;      // flat row in [0, ROWS_)
  const int b    = row0 >> 11;           // S_ = 2048
  const int s0   = row0 & (S_ - 1);
  const int n    = lane & 15;
  const int hi   = lane >> 4;
  // q scale = (1/sqrt(64)) * log2(e) -> softmax uses exp2 directly
  const float scale = (which == 0) ? (0.125f * LOG2E_) : 1.0f;

  const float* arow = inp + (size_t)(row0 + n) * D_ + h * DK_;
  v16h a0 = afrag_f32(arow, 0, hi, scale);
  v16h a1 = afrag_f32(arow, 32, hi, scale);

  v8f acc[4] = {};
#pragma unroll
  for (int c = 0; c < 4; ++c) {
    const _Float16* bp = wT + (c * 16 + n) * DK_ + hi * 16;
    acc[c] = wmma16(a0, bfrag_h(bp),      acc[c]);
    acc[c] = wmma16(a1, bfrag_h(bp + 32), acc[c]);
  }

  const size_t bh = (size_t)(b * H_ + h);
  if (which == 2) {  // V transposed: Vt[bh][d][s]
    _Float16* out = Vt + bh * DK_ * S_;
#pragma unroll
    for (int c = 0; c < 4; ++c)
#pragma unroll
      for (int v = 0; v < 8; ++v)
        out[(size_t)(c * 16 + n) * S_ + s0 + v + hi * 8] = (_Float16)acc[c][v];
  } else {           // Q/K: [bh][s][d]
    _Float16* out = ((which == 0) ? Qh : Kh) + (bh * S_ + s0) * DK_;
#pragma unroll
    for (int c = 0; c < 4; ++c)
#pragma unroll
      for (int v = 0; v < 8; ++v)
        out[(size_t)(v + hi * 8) * DK_ + c * 16 + n] = (_Float16)acc[c][v];
  }
}

// ---------------------------------------------------------------------------
// 2) Flash attention: one wave = 16 query rows of one (b,h), 64-key blocks.
//    128 threads/block = 4 independent waves. Online softmax in exp2 domain.
// ---------------------------------------------------------------------------
__global__ void attn_kernel(const _Float16* __restrict__ Qh,
                            const _Float16* __restrict__ Kh,
                            const _Float16* __restrict__ Vt,
                            _Float16* __restrict__ CtxH) {
  __shared__ __align__(32) _Float16 ptile[4][16 * 64];
  const int wave = threadIdx.x >> 5;
  const int lane = threadIdx.x & 31;
  const int tile = blockIdx.x * 4 + wave;        // 0..4095
  const int qt   = tile & (S_ / 16 - 1);         // query tile
  const int bh   = tile >> 7;                    // b*H + h
  const int b    = bh >> 3, h = bh & 7;
  const int n    = lane & 15;
  const int hi   = lane >> 4;
  const int q0   = qt * 16;

  const _Float16* Qb = Qh + (size_t)bh * S_ * DK_;
  const _Float16* Kb = Kh + (size_t)bh * S_ * DK_;
  const _Float16* Vb = Vt + (size_t)bh * DK_ * S_;

  // Q fragments stay in VGPRs for the whole pass (pre-scaled by log2e/8).
  const _Float16* qrow = Qb + (size_t)(q0 + n) * DK_;
  const v16h aq0 = afrag_h(qrow, 0, hi);
  const v16h aq1 = afrag_h(qrow, 32, hi);

  v8f ctx[4] = {};
  float m_i[8], l_i[8];
#pragma unroll
  for (int v = 0; v < 8; ++v) { m_i[v] = -1e30f; l_i[v] = 0.f; }

  _Float16* pt = ptile[wave];

  for (int j = 0; j < S_; j += 64) {
    if (j + 64 < S_) {  // hint: prefetch next K/V blocks
      __builtin_prefetch(Kb + (size_t)(j + 64 + n) * DK_, 0, 1);
      __builtin_prefetch(Vb + (size_t)(n)*S_ + j + 64, 0, 1);
    }

    // ---- scores: s[c] covers key cols j+16c .. j+16c+15 ----
    v8f s[4];
#pragma unroll
    for (int c = 0; c < 4; ++c) {
      const _Float16* kc = Kb + (size_t)(j + c * 16 + n) * DK_ + hi * 16;
      v8f z = {};
      z = wmma16(aq0, bfrag_h(kc),      z);
      s[c] = wmma16(aq1, bfrag_h(kc + 32), z);
    }

    // ---- online softmax over this 64-key block (base-2 domain) ----
#pragma unroll
    for (int v = 0; v < 8; ++v) {
      float mb = fmaxf(fmaxf(s[0][v], s[1][v]), fmaxf(s[2][v], s[3][v]));
#pragma unroll
      for (int m = 8; m >= 1; m >>= 1) mb = fmaxf(mb, __shfl_xor(mb, m, 16));
      const float mnew = fmaxf(m_i[v], mb);
      const float corr = exp2f(m_i[v] - mnew);
      float e[4], rs = 0.f;
#pragma unroll
      for (int c = 0; c < 4; ++c) { e[c] = exp2f(s[c][v] - mnew); rs += e[c]; }
#pragma unroll
      for (int m = 8; m >= 1; m >>= 1) rs += __shfl_xor(rs, m, 16);
      l_i[v] = l_i[v] * corr + rs;
      m_i[v] = mnew;
#pragma unroll
      for (int c = 0; c < 4; ++c) ctx[c][v] *= corr;
      // P in C-layout -> plain 16x64 tile in LDS (f16)
#pragma unroll
      for (int c = 0; c < 4; ++c)
        pt[(v + hi * 8) * 64 + c * 16 + n] = (_Float16)e[c];
    }
    asm volatile("s_wait_dscnt 0" ::: "memory");

    // Re-read P as two 16x32 A-fragments (row m = lane&15)
    const v16h pa0 = afrag_h(pt + n * 64, 0, hi);
    const v16h pa1 = afrag_h(pt + n * 64, 32, hi);

    // ---- ctx += P @ V  (V transposed -> contiguous B-fragments) ----
#pragma unroll
    for (int c = 0; c < 4; ++c) {
      const _Float16* vb = Vb + (size_t)(c * 16 + n) * S_ + j + hi * 16;
      ctx[c] = wmma16(pa0, bfrag_h(vb),      ctx[c]);
      ctx[c] = wmma16(pa1, bfrag_h(vb + 32), ctx[c]);
    }
  }

  // ---- normalize and store ctx as f16 [B,S,D] (heads merged) ----
  _Float16* ob = CtxH + ((size_t)b * S_ + q0) * D_ + h * DK_;
#pragma unroll
  for (int v = 0; v < 8; ++v) {
    const float inv = 1.0f / l_i[v];
    _Float16* orow = ob + (size_t)(v + hi * 8) * D_;
#pragma unroll
    for (int c = 0; c < 4; ++c)
      orow[c * 16 + n] = (_Float16)(ctx[c][v] * inv);
  }
}

// ---------------------------------------------------------------------------
// 3) Generic WMMA GEMM: out = A(f16,[M,K]) * Bt(f16,[N,K])^T + bias (+resid)
//    One wave computes a 16x64 tile; 4 waves/block.
// ---------------------------------------------------------------------------
__global__ void gemm_kernel(const _Float16* __restrict__ A,
                            const _Float16* __restrict__ Bt,
                            const float* __restrict__ bias,
                            const float* __restrict__ resid,
                            float* __restrict__ outF,
                            _Float16* __restrict__ outH,
                            int M, int N, int K, int do_relu) {
  const int wave = threadIdx.x >> 5;
  const int lane = threadIdx.x & 31;
  const int tile = blockIdx.x * 4 + wave;
  const int nct  = N >> 6;
  const int rt   = tile / nct, ct = tile % nct;
  if (rt * 16 >= M) return;  // uniform per wave
  const int row0 = rt * 16, col0 = ct * 64;
  const int n  = lane & 15;
  const int hi = lane >> 4;

  const _Float16* arow = A + (size_t)(row0 + n) * K;
  v8f acc[4] = {};
  for (int k0 = 0; k0 < K; k0 += 32) {
    const v16h a = afrag_h(arow, k0, hi);
#pragma unroll
    for (int c = 0; c < 4; ++c) {
      const _Float16* bp = Bt + (size_t)(col0 + c * 16 + n) * K + k0 + hi * 16;
      acc[c] = wmma16(a, bfrag_h(bp), acc[c]);
    }
  }

#pragma unroll
  for (int c = 0; c < 4; ++c) {
    const int col = col0 + c * 16 + n;
    const float bv = bias ? bias[col] : 0.f;
#pragma unroll
    for (int v = 0; v < 8; ++v) {
      const int row = row0 + v + hi * 8;
      float val = acc[c][v] + bv;
      if (resid) val += resid[(size_t)row * N + col];
      if (do_relu) val = fmaxf(val, 0.f);
      if (outF) outF[(size_t)row * N + col] = val;
      if (outH) outH[(size_t)row * N + col] = (_Float16)val;
    }
  }
}

// ---------------------------------------------------------------------------
// 4) LayerNorm: one wave per row of D elements; dual f32/f16 output.
// ---------------------------------------------------------------------------
__global__ void layernorm_kernel(const float* __restrict__ x,
                                 const float* __restrict__ g,
                                 const float* __restrict__ be,
                                 float* __restrict__ outF,
                                 _Float16* __restrict__ outH,
                                 int M, int D) {
  const int wave = threadIdx.x >> 5;
  const int lane = threadIdx.x & 31;
  const int row  = blockIdx.x * (blockDim.x >> 5) + wave;
  if (row >= M) return;
  const float* xr = x + (size_t)row * D;
  float s = 0.f, s2 = 0.f;
  for (int i = lane; i < D; i += 32) {
    const float v = xr[i];
    s += v; s2 += v * v;
  }
#pragma unroll
  for (int m = 16; m >= 1; m >>= 1) {
    s  += __shfl_xor(s,  m, 32);
    s2 += __shfl_xor(s2, m, 32);
  }
  const float mu  = s / D;
  const float var = s2 / D - mu * mu;
  const float inv = rsqrtf(var + 1e-5f);
  for (int i = lane; i < D; i += 32) {
    const float v = (xr[i] - mu) * inv * g[i] + be[i];
    if (outF) outF[(size_t)row * D + i] = v;
    if (outH) outH[(size_t)row * D + i] = (_Float16)v;
  }
}

// ---------------------------------------------------------------------------
// 5) Weight transpose + f16 convert: wT[n*K+k] = (f16)w[k*N+n]
// ---------------------------------------------------------------------------
__global__ void transpose_f16_kernel(const float* __restrict__ w,
                                     _Float16* __restrict__ wT,
                                     int K, int N) {
  const int idx = blockIdx.x * blockDim.x + threadIdx.x;
  if (idx < K * N) {
    const int k = idx / N, nn = idx % N;
    wT[(size_t)nn * K + k] = (_Float16)w[idx];
  }
}

// ---------------------------------------------------------------------------
// Host launch
// ---------------------------------------------------------------------------
static inline void* wsoff(void* ws, size_t& off, size_t bytes) {
  void* p = (char*)ws + off;
  off += (bytes + 255) & ~(size_t)255;
  return p;
}

extern "C" void kernel_launch(void* const* d_in, const int* in_sizes, int n_in,
                              void* d_out, int out_size, void* d_ws, size_t ws_size,
                              hipStream_t stream) {
  const float* query = (const float*)d_in[0];
  const float* key   = (const float*)d_in[1];
  const float* value = (const float*)d_in[2];
  const float* wq    = (const float*)d_in[3];
  const float* wk    = (const float*)d_in[4];
  const float* wv    = (const float*)d_in[5];
  const float* wo    = (const float*)d_in[6];
  const float* bo    = (const float*)d_in[7];
  const float* w1    = (const float*)d_in[8];
  const float* b1    = (const float*)d_in[9];
  const float* w2    = (const float*)d_in[10];
  const float* b2    = (const float*)d_in[11];
  const float* g1    = (const float*)d_in[12];
  const float* be1   = (const float*)d_in[13];
  const float* g2    = (const float*)d_in[14];
  const float* be2   = (const float*)d_in[15];

  size_t off = 0;
  _Float16* Qh   = (_Float16*)wsoff(d_ws, off, (size_t)B_ * H_ * S_ * DK_ * 2);
  _Float16* Kh   = (_Float16*)wsoff(d_ws, off, (size_t)B_ * H_ * S_ * DK_ * 2);
  _Float16* Vt   = (_Float16*)wsoff(d_ws, off, (size_t)B_ * H_ * DK_ * S_ * 2);
  _Float16* CtxH = (_Float16*)wsoff(d_ws, off, (size_t)ROWS_ * D_ * 2);
  _Float16* woT  = (_Float16*)wsoff(d_ws, off, (size_t)D_ * D_ * 2);
  _Float16* w1T  = (_Float16*)wsoff(d_ws, off, (size_t)D_ * DFF_ * 2);
  _Float16* w2T  = (_Float16*)wsoff(d_ws, off, (size_t)DFF_ * D_ * 2);
  float*    xpre = (float*)   wsoff(d_ws, off, (size_t)ROWS_ * D_ * 4);
  float*    xln  = (float*)   wsoff(d_ws, off, (size_t)ROWS_ * D_ * 4);
  _Float16* xh   = (_Float16*)wsoff(d_ws, off, (size_t)ROWS_ * D_ * 2);
  _Float16* hh   = (_Float16*)wsoff(d_ws, off, (size_t)ROWS_ * DFF_ * 2);
  float*    ypre = (float*)   wsoff(d_ws, off, (size_t)ROWS_ * D_ * 4);

  // weights -> transposed f16
  transpose_f16_kernel<<<(D_ * D_) / 256, 256, 0, stream>>>(wo, woT, D_, D_);
  transpose_f16_kernel<<<(D_ * DFF_) / 256, 256, 0, stream>>>(w1, w1T, D_, DFF_);
  transpose_f16_kernel<<<(DFF_ * D_) / 256, 256, 0, stream>>>(w2, w2T, DFF_, D_);

  // QKV projections (8 heads = 8 waves per block; y: q/k/v)
  qkv_proj_kernel<<<dim3(ROWS_ / 16, 3), 256, 0, stream>>>(
      query, key, value, wq, wk, wv, Qh, Kh, Vt);

  // attention: 4096 (b,h,qtile) wave-tiles, 4 waves per block
  attn_kernel<<<(B_ * H_ * (S_ / 16)) / 4, 128, 0, stream>>>(Qh, Kh, Vt, CtxH);

  // x_pre = ctx @ wo + bo + query
  gemm_kernel<<<(ROWS_ / 16) * (D_ / 64) / 4, 128, 0, stream>>>(
      CtxH, woT, bo, query, xpre, nullptr, ROWS_, D_, D_, 0);

  // x = LN(x_pre)  -> xln (f32 residual) + xh (f16 GEMM input)
  layernorm_kernel<<<ROWS_ / 8, 256, 0, stream>>>(xpre, g1, be1, xln, xh, ROWS_, D_);

  // h = relu(x @ w1 + b1)  (f16)
  gemm_kernel<<<(ROWS_ / 16) * (DFF_ / 64) / 4, 128, 0, stream>>>(
      xh, w1T, b1, nullptr, nullptr, hh, ROWS_, DFF_, D_, 1);

  // y_pre = h @ w2 + b2 + xln
  gemm_kernel<<<(ROWS_ / 16) * (D_ / 64) / 4, 128, 0, stream>>>(
      hh, w2T, b2, xln, ypre, nullptr, ROWS_, D_, DFF_, 0);

  // out = LN(y_pre)
  layernorm_kernel<<<ROWS_ / 8, 256, 0, stream>>>(ypre, g2, be2, (float*)d_out,
                                                  nullptr, ROWS_, D_);
}
